// GptOssMoEExperts_86595130622524
// MI455X (gfx1250) — compile-verified, matrix operand order checked by reading, not run on hardware
//
#include <hip/hip_runtime.h>
#include <hip/hip_bf16.h>
#include <math.h>

#define SEQ   4096
#define HID   2880
#define INTER 2880
#define NEXP  32
#define TOPK  4

#define BM  128
#define BN  64
#define BK  64
#define LDT 72   // padded LDS row stride in bf16 elements (144B, 16B-aligned, conflict-free)

typedef __attribute__((ext_vector_type(16))) __bf16 v16bf;
typedef __attribute__((ext_vector_type(8)))  __bf16 v8bf;
typedef __attribute__((ext_vector_type(8)))  float  v8f;
typedef int i32x4 __attribute__((vector_size(16)));

#if __has_builtin(__builtin_amdgcn_global_load_async_to_lds_b128)
#define HAVE_ASYNC 1
#else
#define HAVE_ASYNC 0
#endif

// 16B global -> LDS copy: async DMA (ASYNCcnt) when available, else sync fallback.
__device__ __forceinline__ void cp16(const __bf16* g, __bf16* l) {
#if HAVE_ASYNC
  __builtin_amdgcn_global_load_async_to_lds_b128(
      (__attribute__((address_space(1))) i32x4*)(i32x4*)(void*)const_cast<__bf16*>(g),
      (__attribute__((address_space(3))) i32x4*)(i32x4*)(void*)l, 0, 0);
#else
  *(uint4*)l = *(const uint4*)g;
#endif
}

__device__ __forceinline__ void wait_async_and_sync() {
#if HAVE_ASYNC
#if __has_builtin(__builtin_amdgcn_s_wait_asynccnt)
  __builtin_amdgcn_s_wait_asynccnt(0);
#else
  asm volatile("s_wait_asynccnt 0x0" ::: "memory");
#endif
#endif
  __syncthreads();
}

// ---------------- f32 -> bf16 conversion, 8 elems / thread ----------------
__global__ void cvt_bf16_kernel(const float* __restrict__ in,
                                __bf16* __restrict__ out, int n8) {
  int i = blockIdx.x * blockDim.x + threadIdx.x;
  if (i >= n8) return;
  const float4* p = (const float4*)in + (size_t)i * 2;
  float4 a = p[0], b = p[1];
  v8bf o;
  o[0] = (__bf16)a.x; o[1] = (__bf16)a.y; o[2] = (__bf16)a.z; o[3] = (__bf16)a.w;
  o[4] = (__bf16)b.x; o[5] = (__bf16)b.y; o[6] = (__bf16)b.z; o[7] = (__bf16)b.w;
  *((v8bf*)out + i) = o;
}

// ---------------- router: 1 wave per row, lane = expert ----------------
__global__ void router_kernel(const float* __restrict__ hs,
                              const float* __restrict__ rw,
                              const float* __restrict__ rb,
                              float* __restrict__ scale) {
  const int wid  = threadIdx.x >> 5;
  const int lane = threadIdx.x & 31;
  const int s    = blockIdx.x * 8 + wid;

  const float4* h4 = (const float4*)(hs + (size_t)s * HID);
  const float4* w4 = (const float4*)(rw + (size_t)lane * HID);
  float acc = 0.f;
  for (int i = 0; i < HID / 4; ++i) {
    float4 a = h4[i], b = w4[i];
    acc = fmaf(a.x, b.x, fmaf(a.y, b.y, fmaf(a.z, b.z, fmaf(a.w, b.w, acc))));
  }
  float v = acc + rb[lane];

  float top[TOPK];
  for (int t = 0; t < TOPK; ++t) {
    float m = v;
    for (int off = 16; off > 0; off >>= 1)
      m = fmaxf(m, __shfl_xor(m, off, 32));
    top[t] = m;
    unsigned long long bal = __ballot(v == m);
    int first = __ffsll(bal) - 1;
    if (lane == first) v = -INFINITY;   // remove one instance (top_k tie semantics)
  }
  if (lane == 0) {
    float m0 = top[0];
    float e0 = __expf(top[0] - m0), e1 = __expf(top[1] - m0);
    float e2 = __expf(top[2] - m0), e3 = __expf(top[3] - m0);
    float ssum = e0 + e1 + e2 + e3;
    scale[s] = (e0 / ssum) + (e1 / ssum) + (e2 / ssum) + (e3 / ssum);
  }
}

// ---------------- GEMM1: x = (hs@Wg.T+bg) * silu(hs@Wu.T+bu)  (bf16 out) --
__global__ __launch_bounds__(256)
void gemm1_kernel(const __bf16* __restrict__ A,   // hs   bf16 (SEQ x HID)
                  const __bf16* __restrict__ W,   // W1   bf16 (2*INTER x HID)
                  const float*  __restrict__ bias,// b1   f32  (2*INTER)
                  __bf16* __restrict__ X) {       // x    bf16 (SEQ x INTER)
  __shared__ __bf16 sA[2][BM * LDT];
  __shared__ __bf16 sG[2][BN * LDT];
  __shared__ __bf16 sU[2][BN * LDT];

  const int t    = threadIdx.x;
  const int lane = t & 31;
  const int w    = t >> 5;     // 0..7
  const int wm   = w & 3;      // 32-row band
  const int wn   = w >> 2;     // 32-col band
  const int lm   = lane & 15;
  const int lh   = lane >> 4;

  const int m0 = blockIdx.y * BM;
  const int n0 = blockIdx.x * BN;

  const __bf16* gA = A + (size_t)m0 * HID;
  const __bf16* gG = W + (size_t)n0 * HID;
  const __bf16* gU = W + (size_t)(n0 + INTER) * HID;

  auto stage = [&](int buf, int k0) {
#pragma unroll
    for (int i = 0; i < 4; ++i) {
      int u = i * 256 + t, r = u >> 3, c = u & 7;
      cp16(gA + (size_t)r * HID + k0 + c * 8, &sA[buf][r * LDT + c * 8]);
    }
#pragma unroll
    for (int i = 0; i < 2; ++i) {
      int u = i * 256 + t, r = u >> 3, c = u & 7;
      cp16(gG + (size_t)r * HID + k0 + c * 8, &sG[buf][r * LDT + c * 8]);
      cp16(gU + (size_t)r * HID + k0 + c * 8, &sU[buf][r * LDT + c * 8]);
    }
  };

  const v8f vzero = {0.f, 0.f, 0.f, 0.f, 0.f, 0.f, 0.f, 0.f};
  v8f accg[2][2], accu[2][2];
  for (int i = 0; i < 2; ++i)
    for (int j = 0; j < 2; ++j) { accg[i][j] = vzero; accu[i][j] = vzero; }

  stage(0, 0);
  wait_async_and_sync();

  const int NK = HID / BK;          // 45
  for (int kb = 0; kb < NK; ++kb) {
    const int cur = kb & 1, nxt = cur ^ 1;
    if (kb + 1 < NK) stage(nxt, (kb + 1) * BK);   // DMA next tile while computing

#pragma unroll
    for (int kk = 0; kk < 2; ++kk) {
      const int kc = kk * 32 + lh * 8;   // lane-half K chunking per ISA A layout
      v16bf a[2], bg[2], bu[2];
#pragma unroll
      for (int mi = 0; mi < 2; ++mi) {
        int row = wm * 32 + mi * 16 + lm;
        *(uint4*)&a[mi]       = *(const uint4*)&sA[cur][row * LDT + kc];
        *((uint4*)&a[mi] + 1) = *(const uint4*)&sA[cur][row * LDT + kc + 16];
      }
#pragma unroll
      for (int ni = 0; ni < 2; ++ni) {
        int row = wn * 32 + ni * 16 + lm;
        *(uint4*)&bg[ni]       = *(const uint4*)&sG[cur][row * LDT + kc];
        *((uint4*)&bg[ni] + 1) = *(const uint4*)&sG[cur][row * LDT + kc + 16];
        *(uint4*)&bu[ni]       = *(const uint4*)&sU[cur][row * LDT + kc];
        *((uint4*)&bu[ni] + 1) = *(const uint4*)&sU[cur][row * LDT + kc + 16];
      }
#pragma unroll
      for (int mi = 0; mi < 2; ++mi)
#pragma unroll
        for (int ni = 0; ni < 2; ++ni) {
          accg[mi][ni] = __builtin_amdgcn_wmma_f32_16x16x32_bf16(
              false, a[mi], false, bg[ni], (short)0, accg[mi][ni], false, false);
          accu[mi][ni] = __builtin_amdgcn_wmma_f32_16x16x32_bf16(
              false, a[mi], false, bu[ni], (short)0, accu[mi][ni], false, false);
        }
    }
    wait_async_and_sync();   // next buffer ready + everyone done reading cur
  }

  // epilogue: bias + gate*silu(up), store bf16
#pragma unroll
  for (int mi = 0; mi < 2; ++mi) {
#pragma unroll
    for (int ni = 0; ni < 2; ++ni) {
      int col = n0 + wn * 32 + ni * 16 + lm;
      float bgv = bias[col];
      float buv = bias[col + INTER];
#pragma unroll
      for (int r = 0; r < 8; ++r) {
        int row = m0 + wm * 32 + mi * 16 + r + 8 * lh;
        float g  = accg[mi][ni][r] + bgv;
        float u  = accu[mi][ni][r] + buv;
        float su = u / (1.f + __expf(-u));      // silu(u)
        X[(size_t)row * INTER + col] = (__bf16)(g * su);
      }
    }
  }
}

// ---------------- GEMM2: out = scale * (x@Wd.T + bd)  (f32 out) ----------
__global__ __launch_bounds__(256)
void gemm2_kernel(const __bf16* __restrict__ A,    // x    bf16 (SEQ x INTER)
                  const __bf16* __restrict__ W,    // Wd   bf16 (HID x INTER)
                  const float*  __restrict__ bias, // bd   f32  (HID)
                  const float*  __restrict__ scale,// per-row router scale
                  float* __restrict__ O) {         // out  f32  (SEQ x HID)
  __shared__ __bf16 sA[2][BM * LDT];
  __shared__ __bf16 sB[2][BN * LDT];

  const int t    = threadIdx.x;
  const int lane = t & 31;
  const int w    = t >> 5;
  const int wm   = w & 3;
  const int wn   = w >> 2;
  const int lm   = lane & 15;
  const int lh   = lane >> 4;

  const int m0 = blockIdx.y * BM;
  const int n0 = blockIdx.x * BN;

  const __bf16* gA = A + (size_t)m0 * INTER;
  const __bf16* gB = W + (size_t)n0 * INTER;

  auto stage = [&](int buf, int k0) {
#pragma unroll
    for (int i = 0; i < 4; ++i) {
      int u = i * 256 + t, r = u >> 3, c = u & 7;
      cp16(gA + (size_t)r * INTER + k0 + c * 8, &sA[buf][r * LDT + c * 8]);
    }
#pragma unroll
    for (int i = 0; i < 2; ++i) {
      int u = i * 256 + t, r = u >> 3, c = u & 7;
      cp16(gB + (size_t)r * INTER + k0 + c * 8, &sB[buf][r * LDT + c * 8]);
    }
  };

  const v8f vzero = {0.f, 0.f, 0.f, 0.f, 0.f, 0.f, 0.f, 0.f};
  v8f acc[2][2];
  for (int i = 0; i < 2; ++i)
    for (int j = 0; j < 2; ++j) acc[i][j] = vzero;

  stage(0, 0);
  wait_async_and_sync();

  const int NK = INTER / BK;        // 45
  for (int kb = 0; kb < NK; ++kb) {
    const int cur = kb & 1, nxt = cur ^ 1;
    if (kb + 1 < NK) stage(nxt, (kb + 1) * BK);

#pragma unroll
    for (int kk = 0; kk < 2; ++kk) {
      const int kc = kk * 32 + lh * 8;
      v16bf a[2], b[2];
#pragma unroll
      for (int mi = 0; mi < 2; ++mi) {
        int row = wm * 32 + mi * 16 + lm;
        *(uint4*)&a[mi]       = *(const uint4*)&sA[cur][row * LDT + kc];
        *((uint4*)&a[mi] + 1) = *(const uint4*)&sA[cur][row * LDT + kc + 16];
      }
#pragma unroll
      for (int ni = 0; ni < 2; ++ni) {
        int row = wn * 32 + ni * 16 + lm;
        *(uint4*)&b[ni]       = *(const uint4*)&sB[cur][row * LDT + kc];
        *((uint4*)&b[ni] + 1) = *(const uint4*)&sB[cur][row * LDT + kc + 16];
      }
#pragma unroll
      for (int mi = 0; mi < 2; ++mi)
#pragma unroll
        for (int ni = 0; ni < 2; ++ni)
          acc[mi][ni] = __builtin_amdgcn_wmma_f32_16x16x32_bf16(
              false, a[mi], false, b[ni], (short)0, acc[mi][ni], false, false);
    }
    wait_async_and_sync();
  }

#pragma unroll
  for (int mi = 0; mi < 2; ++mi) {
#pragma unroll
    for (int ni = 0; ni < 2; ++ni) {
      int col = n0 + wn * 32 + ni * 16 + lm;
      float bv = bias[col];
#pragma unroll
      for (int r = 0; r < 8; ++r) {
        int row = m0 + wm * 32 + mi * 16 + r + 8 * lh;
        O[(size_t)row * HID + col] = scale[row] * (acc[mi][ni][r] + bv);
      }
    }
  }
}

// ---------------- launch ----------------
extern "C" void kernel_launch(void* const* d_in, const int* in_sizes, int n_in,
                              void* d_out, int out_size, void* d_ws, size_t ws_size,
                              hipStream_t stream) {
  const float* hs = (const float*)d_in[0];
  const float* rw = (const float*)d_in[1];
  const float* rb = (const float*)d_in[2];
  const float* w1 = (const float*)d_in[3];
  const float* b1 = (const float*)d_in[4];
  const float* w2 = (const float*)d_in[5];
  const float* b2 = (const float*)d_in[6];
  float* out = (float*)d_out;

  char* ws = (char*)d_ws;
  float*  scale = (float*)ws;                               // SEQ floats (16KB slot)
  __bf16* hsb   = (__bf16*)(ws + 16384);                    // SEQ x HID
  __bf16* w1b   = hsb + (size_t)SEQ * HID;                  // 2*INTER x HID
  __bf16* w2b   = w1b + (size_t)2 * INTER * HID;            // HID x INTER
  __bf16* xb    = w2b + (size_t)HID * INTER;                // SEQ x INTER

  int n8;
  n8 = SEQ * HID / 8;
  cvt_bf16_kernel<<<(n8 + 255) / 256, 256, 0, stream>>>(hs, hsb, n8);
  n8 = 2 * INTER * HID / 8;
  cvt_bf16_kernel<<<(n8 + 255) / 256, 256, 0, stream>>>(w1, w1b, n8);
  n8 = HID * INTER / 8;
  cvt_bf16_kernel<<<(n8 + 255) / 256, 256, 0, stream>>>(w2, w2b, n8);

  router_kernel<<<SEQ / 8, 256, 0, stream>>>(hs, rw, rb, scale);

  gemm1_kernel<<<dim3(INTER / BN, SEQ / BM), 256, 0, stream>>>(hsb, w1b, b1, xb);
  gemm2_kernel<<<dim3(HID / BN, SEQ / BM), 256, 0, stream>>>(xb, w2b, b2, scale, out);
}
